// RelMultiHeadedAttention_87677462381045
// MI455X (gfx1250) — matrix operand order repaired and checked by reading, not compiled, not run on hardware
//
#include <hip/hip_runtime.h>
#include <hip/hip_bf16.h>
#include <stdint.h>

// ---------------------------------------------------------------------------
// Problem constants (Transformer-XL relative attention)
// B=2 S=1024 M=1024 D=1024 H=16 DH=64 K=2048 R=4095
// ---------------------------------------------------------------------------
#define CB 2
#define CS 1024
#define CM 1024
#define CD 1024
#define CH 16
#define CDH 64
#define CK 2048
#define CR 4095

typedef __attribute__((ext_vector_type(16))) __bf16 v16bf;
typedef __attribute__((ext_vector_type(8)))  float  v8f;

union FragU {
    v16bf v;
    uint4 u[2];
};

// ---------------------------------------------------------------------------
// WMMA fragment loaders (per CDNA5 ISA VGPR layouts, wave32)
// A 16x32 bf16 (MxK): lane L: row = L&15 ; kb=(L>>4)*8 -> halves
//   [kb..kb+7] and [kb+16..kb+23] (two 16B chunks).
// B 32x16 bf16 (KxN): lane L: col = L&15 ; kb=(L>>4)*8 -> halves
//   [kb..kb+7] and [kb+16..kb+23] (N-major table: row=n, halves=k).
// C/D 16x16 f32: lane L: col = L&15 ; elem v: row = v + ((L>>4)<<3).
// ---------------------------------------------------------------------------
__device__ __forceinline__ v16bf load_fragA(const __bf16* base, int row0, int ld, int k0) {
    const int l = threadIdx.x & 31;
    const __bf16* p = base + (size_t)(row0 + (l & 15)) * ld + k0 + ((l >> 4) << 3);
    FragU f;
    f.u[0] = *reinterpret_cast<const uint4*>(p);
    f.u[1] = *reinterpret_cast<const uint4*>(p + 16);
    return f.v;
}

__device__ __forceinline__ v16bf load_fragB(const __bf16* base, int n0, int ld, int k0) {
    const int l = threadIdx.x & 31;
    const __bf16* p = base + (size_t)(n0 + (l & 15)) * ld + k0 + ((l >> 4) << 3);
    FragU f;
    f.u[0] = *reinterpret_cast<const uint4*>(p);
    f.u[1] = *reinterpret_cast<const uint4*>(p + 16);
    return f.v;
}

#define WMMA_BF16(A, Bm, C) \
    __builtin_amdgcn_wmma_f32_16x16x32_bf16(false, (A), false, (Bm), (short)0, (C), false, false)

// ---------------------------------------------------------------------------
// Async global->LDS copies (CDNA5, tracked by ASYNCcnt).
// 128-thread block; each lane copies 16B per instruction.
// ---------------------------------------------------------------------------
__device__ __forceinline__ void async_tile_linear(const __bf16* g, uint32_t lds) {
    const int t = threadIdx.x;                      // 0..127, 512 chunks of 16B
#pragma unroll
    for (int i = 0; i < 4; ++i) {
        const int c = t + i * 128;
        const unsigned long long ga = (unsigned long long)(uintptr_t)g + (unsigned long long)c * 16u;
        const uint32_t la = lds + (uint32_t)c * 16u;
        asm volatile("global_load_async_to_lds_b128 %0, %1, off"
                     :: "v"(la), "v"(ga) : "memory");
    }
}

// V is stored transposed (DH=64 rows, ld = CK halves); copy 64x64 tile at jj0.
__device__ __forceinline__ void async_tile_v(const __bf16* vbh, int jj0, uint32_t lds) {
    const int t = threadIdx.x;
#pragma unroll
    for (int i = 0; i < 4; ++i) {
        const int c    = t + i * 128;
        const int dh   = c >> 3;
        const int off8 = (c & 7) * 8;
        const unsigned long long ga =
            (unsigned long long)(uintptr_t)(vbh + (size_t)dh * CK + jj0 + off8);
        const uint32_t la = lds + (uint32_t)((dh * 64 + off8) * 2);
        asm volatile("global_load_async_to_lds_b128 %0, %1, off"
                     :: "v"(la), "v"(ga) : "memory");
    }
}

// ---------------------------------------------------------------------------
// Kernel 1: fused LayerNorm of concat(memory, hidden).
// ---------------------------------------------------------------------------
__global__ void ln_kernel(const float* __restrict__ hidden, const float* __restrict__ memory,
                          const float* __restrict__ gamma, const float* __restrict__ beta,
                          __bf16* __restrict__ cb, float* __restrict__ hln) {
    const int row  = blockIdx.x;          // 0 .. B*K-1
    const int b    = row >> 11;
    const int tpos = row & 2047;
    const float* src = (tpos < CM)
        ? (memory + ((size_t)b * CM + tpos) * CD)
        : (hidden + ((size_t)b * CS + (tpos - CM)) * CD);
    const int tid = threadIdx.x;

    float x[4];
    float s = 0.f;
#pragma unroll
    for (int j = 0; j < 4; ++j) { x[j] = src[tid + j * 256]; s += x[j]; }

    __shared__ float red[256];
    red[tid] = s;
    __syncthreads();
    for (int w = 128; w > 0; w >>= 1) {
        if (tid < w) red[tid] += red[tid + w];
        __syncthreads();
    }
    const float mu = red[0] * (1.0f / (float)CD);
    __syncthreads();

    float q = 0.f;
#pragma unroll
    for (int j = 0; j < 4; ++j) { float d = x[j] - mu; q += d * d; }
    red[tid] = q;
    __syncthreads();
    for (int w = 128; w > 0; w >>= 1) {
        if (tid < w) red[tid] += red[tid + w];
        __syncthreads();
    }
    const float var  = red[0] * (1.0f / (float)CD);
    const float rstd = rsqrtf(var + 1e-5f);

    __bf16* crow = cb + (size_t)row * CD;
    float*  hrow = (tpos >= CM) ? (hln + ((size_t)b * CS + (tpos - CM)) * CD) : nullptr;
#pragma unroll
    for (int j = 0; j < 4; ++j) {
        const int d  = tid + j * 256;
        const float y = (x[j] - mu) * rstd * gamma[d] + beta[d];
        crow[d] = (__bf16)y;
        if (hrow) hrow[d] = y;
    }
}

// ---------------------------------------------------------------------------
// Kernel 2: transpose-cast 1024x1024 f32 weight -> bf16 N-major.
// ---------------------------------------------------------------------------
__global__ void cast_transpose_w(const float* __restrict__ W, __bf16* __restrict__ Wt) {
    const int idx = blockIdx.x * 256 + threadIdx.x;
    const int k = idx >> 10;
    const int n = idx & 1023;
    Wt[(size_t)n * CD + k] = (__bf16)W[idx];
}

__global__ void cast_pos(const float* __restrict__ P, __bf16* __restrict__ Pb, int n) {
    const int i = blockIdx.x * 256 + threadIdx.x;
    if (i < n) Pb[i] = (__bf16)P[i];
}

// ---------------------------------------------------------------------------
// Kernel 3: WMMA GEMM  C[m,n] = A[m,:1024] @ W[:1024,n]
// Block = 4 waves; wave tile = 32(M) x 64(N) (8 accumulators), software-
// pipelined k-loop (prefetch k+32 fragments while computing k).
// modes: 0 Q->(B,H,S,DH) ; 1 K->(B,H,K,DH) ; 2 V->(B,H,DH,K) ;
//        3 output projection: f32 = acc + resid.
// ---------------------------------------------------------------------------
__global__ void __launch_bounds__(128)
gemm_wmma(const __bf16* __restrict__ A, const __bf16* __restrict__ Wt,
          int rowsPerBatch, int aBatchStride, int rowOffset, int mode,
          __bf16* __restrict__ outB, float* __restrict__ outF,
          const float* __restrict__ resid) {
    const int wave = threadIdx.x >> 5;
    const int l    = threadIdx.x & 31;
    const int rowbase = blockIdx.x * 128 + wave * 32;
    const int n0      = blockIdx.y * 64;
    const int b    = rowbase / rowsPerBatch;
    const int pos0 = rowbase % rowsPerBatch;
    const __bf16* Arow = A + (size_t)(b * aBatchStride + rowOffset + pos0) * CD;

    const v8f z8 = {0.f, 0.f, 0.f, 0.f, 0.f, 0.f, 0.f, 0.f};
    v8f acc[8] = {z8, z8, z8, z8, z8, z8, z8, z8};

    v16bf cA0 = load_fragA(Arow, 0,  CD, 0);
    v16bf cA1 = load_fragA(Arow, 16, CD, 0);
    v16bf cB0 = load_fragB(Wt, n0 +  0, CD, 0);
    v16bf cB1 = load_fragB(Wt, n0 + 16, CD, 0);
    v16bf cB2 = load_fragB(Wt, n0 + 32, CD, 0);
    v16bf cB3 = load_fragB(Wt, n0 + 48, CD, 0);

    for (int kk = 0; kk + 32 < CD; kk += 32) {
        const int kn = kk + 32;
        // prefetch next k-step
        const v16bf nA0 = load_fragA(Arow, 0,  CD, kn);
        const v16bf nA1 = load_fragA(Arow, 16, CD, kn);
        const v16bf nB0 = load_fragB(Wt, n0 +  0, CD, kn);
        const v16bf nB1 = load_fragB(Wt, n0 + 16, CD, kn);
        const v16bf nB2 = load_fragB(Wt, n0 + 32, CD, kn);
        const v16bf nB3 = load_fragB(Wt, n0 + 48, CD, kn);
        // compute current k-step
        acc[0] = WMMA_BF16(cA0, cB0, acc[0]);
        acc[1] = WMMA_BF16(cA0, cB1, acc[1]);
        acc[2] = WMMA_BF16(cA0, cB2, acc[2]);
        acc[3] = WMMA_BF16(cA0, cB3, acc[3]);
        acc[4] = WMMA_BF16(cA1, cB0, acc[4]);
        acc[5] = WMMA_BF16(cA1, cB1, acc[5]);
        acc[6] = WMMA_BF16(cA1, cB2, acc[6]);
        acc[7] = WMMA_BF16(cA1, cB3, acc[7]);
        cA0 = nA0; cA1 = nA1; cB0 = nB0; cB1 = nB1; cB2 = nB2; cB3 = nB3;
    }
    // epilogue k-step
    acc[0] = WMMA_BF16(cA0, cB0, acc[0]);
    acc[1] = WMMA_BF16(cA0, cB1, acc[1]);
    acc[2] = WMMA_BF16(cA0, cB2, acc[2]);
    acc[3] = WMMA_BF16(cA0, cB3, acc[3]);
    acc[4] = WMMA_BF16(cA1, cB0, acc[4]);
    acc[5] = WMMA_BF16(cA1, cB1, acc[5]);
    acc[6] = WMMA_BF16(cA1, cB2, acc[6]);
    acc[7] = WMMA_BF16(cA1, cB3, acc[7]);

    const int colL  = l & 15;
    const int half8 = (l >> 4) << 3;
#pragma unroll
    for (int mt = 0; mt < 2; ++mt) {
#pragma unroll
        for (int nt = 0; nt < 4; ++nt) {
#pragma unroll
            for (int v = 0; v < 8; ++v) {
                const int pos = pos0 + mt * 16 + v + half8;
                const int n   = n0 + nt * 16 + colL;
                const float val = acc[mt * 4 + nt][v];
                if (mode == 3) {
                    const size_t o = ((size_t)(b * rowsPerBatch + pos)) * CD + n;
                    outF[o] = val + resid[o];
                } else {
                    const int h = n >> 6, dh = n & 63;
                    size_t o;
                    if (mode == 0)      o = (((size_t)(b * CH + h)) * CS + pos) * CDH + dh;
                    else if (mode == 1) o = (((size_t)(b * CH + h)) * CK + pos) * CDH + dh;
                    else                o = (((size_t)(b * CH + h)) * CDH + dh) * CK + pos;
                    outB[o] = (__bf16)val;
                }
            }
        }
    }
}

// ---------------------------------------------------------------------------
// Kernel 4: fused relative attention (flash style, online softmax).
// Grid (S/64, H, B), block = 128 (4 waves), wave owns 16 query rows.
// K/V 64x64 tiles are DMA'd to double-buffered LDS with
// GLOBAL_LOAD_ASYNC_TO_LDS_B128 (ASYNCcnt) and shared by all 4 waves.
// Per key tile: Sc=Q@K^T (8 WMMA), T=Q@pos_window^T (10 WMMA, skew gather
// T[row, col-row+15]), online softmax with causal mask, O += P@V (8 WMMA).
// ---------------------------------------------------------------------------
__global__ void __launch_bounds__(128)
flash_attn(const __bf16* __restrict__ Q, const __bf16* __restrict__ Kc,
           const __bf16* __restrict__ Vt, const __bf16* __restrict__ Pos,
           __bf16* __restrict__ attnb) {
    const int wave = threadIdx.x >> 5;
    const int l    = threadIdx.x & 31;
    const int i0   = blockIdx.x * 64;
    const int iw   = i0 + wave * 16;
    const int h    = blockIdx.y;
    const int b    = blockIdx.z;

    const __bf16* Qbh = Q   + ((size_t)(b * CH + h)) * CS  * CDH;
    const __bf16* Kbh = Kc  + ((size_t)(b * CH + h)) * CK  * CDH;
    const __bf16* Vbh = Vt  + ((size_t)(b * CH + h)) * CDH * CK;
    const __bf16* Pb  = Pos + (size_t)b * CR * CDH;

    __shared__ __align__(16) __bf16 ldsK[2][64 * 64];   // 16 KB
    __shared__ __align__(16) __bf16 ldsV[2][64 * 64];   // 16 KB
    __shared__ __align__(16) __bf16 Plds[4][16 * 64];   //  8 KB
    __shared__ __align__(16) float  Tlds[4][16 * 80];   // 20 KB

    const uint32_t aK[2] = { (uint32_t)(uintptr_t)&ldsK[0][0], (uint32_t)(uintptr_t)&ldsK[1][0] };
    const uint32_t aV[2] = { (uint32_t)(uintptr_t)&ldsV[0][0], (uint32_t)(uintptr_t)&ldsV[1][0] };

    const v16bf qf0 = load_fragA(Qbh, iw, CDH, 0);
    const v16bf qf1 = load_fragA(Qbh, iw, CDH, 32);

    const v8f z8 = {0.f, 0.f, 0.f, 0.f, 0.f, 0.f, 0.f, 0.f};
    v8f O[4] = {z8, z8, z8, z8};
    float m[8], lsum[8];
#pragma unroll
    for (int v = 0; v < 8; ++v) { m[v] = -3.0e38f; lsum[v] = 0.f; }

    const int colL  = l & 15;
    const int half8 = (l >> 4) << 3;
    const int ntile = (i0 + 63 + CM) / 64 + 1;   // uniform per block

    // preload tile 0
    async_tile_linear(Kbh, aK[0]);
    async_tile_v(Vbh, 0, aV[0]);
    asm volatile("s_wait_asynccnt 0" ::: "memory");
    __syncthreads();

    int buf = 0;
    for (int t = 0; t < ntile; ++t) {
        const int jj0 = t * 64;
        if (t + 1 < ntile) {   // prefetch next tile into other buffer
            async_tile_linear(Kbh + (size_t)(jj0 + 64) * CDH, aK[buf ^ 1]);
            async_tile_v(Vbh, jj0 + 64, aV[buf ^ 1]);
        }

        // ---- scores Sc = Q @ K^T (K tile from LDS) ------------------------
        v8f sc[4] = {z8, z8, z8, z8};
#pragma unroll
        for (int nt = 0; nt < 4; ++nt) {
            v16bf bk = load_fragB(&ldsK[buf][0], nt * 16, 64, 0);
            sc[nt] = WMMA_BF16(qf0, bk, sc[nt]);
            bk = load_fragB(&ldsK[buf][0], nt * 16, 64, 32);
            sc[nt] = WMMA_BF16(qf1, bk, sc[nt]);
        }

        // ---- relative position window T = Q @ pos[r0..r0+79]^T ------------
        const int r0 = jj0 - iw + 1008;
        v8f tp[5] = {z8, z8, z8, z8, z8};
#pragma unroll
        for (int nt = 0; nt < 5; ++nt) {
            v16bf bp = load_fragB(Pb, r0 + nt * 16, CDH, 0);
            tp[nt] = WMMA_BF16(qf0, bp, tp[nt]);
            bp = load_fragB(Pb, r0 + nt * 16, CDH, 32);
            tp[nt] = WMMA_BF16(qf1, bp, tp[nt]);
        }
#pragma unroll
        for (int nt = 0; nt < 5; ++nt)
#pragma unroll
            for (int v = 0; v < 8; ++v)
                Tlds[wave][(v + half8) * 80 + nt * 16 + colL] = tp[nt][v];
        asm volatile("s_wait_dscnt 0" ::: "memory");

        // ---- combine + skew gather + causal mask --------------------------
        float sv[4][8];
#pragma unroll
        for (int nt = 0; nt < 4; ++nt)
#pragma unroll
            for (int v = 0; v < 8; ++v) {
                const int ii  = v + half8;
                const int col = nt * 16 + colL;
                const int tt  = col - ii + 15;               // in [0, 79]
                const float ps = Tlds[wave][ii * 80 + tt];
                float s = (sc[nt][v] + ps) * 0.03125f;       // 1/sqrt(1024)
                if (jj0 + col > iw + ii + CM) s = -3.0e38f;  // causal mask
                sv[nt][v] = s;
            }

        // ---- online softmax -----------------------------------------------
        float mnew[8], rowsum[8];
#pragma unroll
        for (int v = 0; v < 8; ++v) {
            float mx = fmaxf(fmaxf(sv[0][v], sv[1][v]), fmaxf(sv[2][v], sv[3][v]));
            for (int off = 1; off < 16; off <<= 1)
                mx = fmaxf(mx, __shfl_xor(mx, off, 32));
            mnew[v]   = fmaxf(m[v], mx);
            rowsum[v] = 0.f;
        }
#pragma unroll
        for (int nt = 0; nt < 4; ++nt)
#pragma unroll
            for (int v = 0; v < 8; ++v) {
                const float p = __expf(sv[nt][v] - mnew[v]);
                rowsum[v] += p;
                Plds[wave][(v + half8) * 64 + nt * 16 + colL] = (__bf16)p;
            }
#pragma unroll
        for (int v = 0; v < 8; ++v) {
            for (int off = 1; off < 16; off <<= 1)
                rowsum[v] += __shfl_xor(rowsum[v], off, 32);
            const float fac = __expf(m[v] - mnew[v]);
            lsum[v] = lsum[v] * fac + rowsum[v];
            m[v]    = mnew[v];
#pragma unroll
            for (int nt = 0; nt < 4; ++nt) O[nt][v] *= fac;
        }
        asm volatile("s_wait_dscnt 0" ::: "memory");

        // ---- O += P @ V (V tile from LDS, dh-major) -----------------------
        for (int kk = 0; kk < 64; kk += 32) {
            const v16bf pa = load_fragA(&Plds[wave][0], 0, 64, kk);
#pragma unroll
            for (int nt = 0; nt < 4; ++nt) {
                const v16bf bv = load_fragB(&ldsV[buf][0], nt * 16, 64, kk);
                O[nt] = WMMA_BF16(pa, bv, O[nt]);
            }
        }

        asm volatile("s_wait_asynccnt 0" ::: "memory");
        __syncthreads();
        buf ^= 1;
    }

    // ---- epilogue: normalize, store bf16 attention rows (B,S,H*DH) --------
#pragma unroll
    for (int v = 0; v < 8; ++v) {
        const float inv = 1.0f / lsum[v];
        const int s = iw + v + half8;
#pragma unroll
        for (int nt = 0; nt < 4; ++nt) {
            const float val = O[nt][v] * inv;
            const size_t o = ((size_t)(b * CS + s)) * (CH * CDH) + h * CDH + nt * 16 + colL;
            attnb[o] = (__bf16)val;
        }
    }
}

// ---------------------------------------------------------------------------
// Host orchestration. Workspace layout (bytes):
//   cb bf16 @0 (8MB) | hln f32 @8MB (8MB) | Wtq/Wtk/Wtv/Wto bf16 @16MB (2MB ea)
//   posb bf16 @24MB (~1MB) | Qb @26MB (4MB) | Kb @30MB (8MB) | Vtb @38MB (8MB)
//   attn bf16 @46MB (4MB)  -> 50MB total
// ---------------------------------------------------------------------------
extern "C" void kernel_launch(void* const* d_in, const int* in_sizes, int n_in,
                              void* d_out, int out_size, void* d_ws, size_t ws_size,
                              hipStream_t stream) {
    const float* hidden = (const float*)d_in[0];
    const float* posemb = (const float*)d_in[1];
    const float* memory = (const float*)d_in[2];
    // d_in[3] = mask (causality handled analytically)
    const float* Wq = (const float*)d_in[4];
    const float* Wk = (const float*)d_in[5];
    const float* Wv = (const float*)d_in[6];
    const float* Wo = (const float*)d_in[7];
    const float* gamma = (const float*)d_in[8];
    const float* beta  = (const float*)d_in[9];
    float* out = (float*)d_out;

    char* ws = (char*)d_ws;
    const size_t MB = 1024ull * 1024ull;
    __bf16* cb   = (__bf16*)(ws + 0);
    float*  hln  = (float*) (ws + 8 * MB);
    __bf16* Wtq  = (__bf16*)(ws + 16 * MB);
    __bf16* Wtk  = (__bf16*)(ws + 18 * MB);
    __bf16* Wtv  = (__bf16*)(ws + 20 * MB);
    __bf16* Wto  = (__bf16*)(ws + 22 * MB);
    __bf16* posb = (__bf16*)(ws + 24 * MB);
    __bf16* Qb   = (__bf16*)(ws + 26 * MB);
    __bf16* Kb   = (__bf16*)(ws + 30 * MB);
    __bf16* Vtb  = (__bf16*)(ws + 38 * MB);
    __bf16* attn = (__bf16*)(ws + 46 * MB);

    // 1) LayerNorm of concat(memory, hidden)
    ln_kernel<<<dim3(CB * CK), 256, 0, stream>>>(hidden, memory, gamma, beta, cb, hln);

    // 2) weight + pos-emb casts
    cast_transpose_w<<<dim3(4096), 256, 0, stream>>>(Wq, Wtq);
    cast_transpose_w<<<dim3(4096), 256, 0, stream>>>(Wk, Wtk);
    cast_transpose_w<<<dim3(4096), 256, 0, stream>>>(Wv, Wtv);
    cast_transpose_w<<<dim3(4096), 256, 0, stream>>>(Wo, Wto);
    const int nPos = CB * CR * CDH;
    cast_pos<<<dim3((nPos + 255) / 256), 256, 0, stream>>>(posemb, posb, nPos);

    // 3) QKV projections (WMMA, 128 rows per block)
    gemm_wmma<<<dim3(CB * CS / 128, 16), 128, 0, stream>>>(
        cb, Wtq, CS, CK, CM, 0, Qb, nullptr, nullptr);
    gemm_wmma<<<dim3(CB * CK / 128, 16), 128, 0, stream>>>(
        cb, Wtk, CK, CK, 0, 1, Kb, nullptr, nullptr);
    gemm_wmma<<<dim3(CB * CK / 128, 16), 128, 0, stream>>>(
        cb, Wtv, CK, CK, 0, 2, Vtb, nullptr, nullptr);

    // 4) fused relative attention
    flash_attn<<<dim3(CS / 64, CH, CB), 128, 0, stream>>>(Qb, Kb, Vtb, posb, attn);

    // 5) output projection + residual (WMMA)
    gemm_wmma<<<dim3(CB * CS / 128, 16), 128, 0, stream>>>(
        attn, Wto, CS, CS, 0, 3, nullptr, out, hln);
}
